// MoEConnectionProcessor_27900107555503
// MI455X (gfx1250) — matrix operand order repaired and checked by reading, not compiled, not run on hardware
//
#include <hip/hip_runtime.h>
#include <math.h>

#define N_CELLS 19683
#define KNBR 26
#define D 128
#define HG 64
#define DT 0.1f

typedef _Float16 v16h __attribute__((ext_vector_type(16)));
typedef _Float16 v8h  __attribute__((ext_vector_type(8)));
typedef _Float16 v4h  __attribute__((ext_vector_type(4)));
typedef float    v8f  __attribute__((ext_vector_type(8)));
typedef float    v4f  __attribute__((ext_vector_type(4)));

static __device__ inline v16h combine16(v8h lo, v8h hi) {
    return __builtin_shufflevector(lo, hi, 0,1,2,3,4,5,6,7,8,9,10,11,12,13,14,15);
}

// WMMA wrapper: D = A*B + C, f16 in, f32 accum
static __device__ inline v8f wmma_f16(v16h a, v16h b, v8f c) {
    return __builtin_amdgcn_wmma_f32_16x16x32_f16(false, a, false, b, (short)0, c, false, false);
}

// ---- A-matrix fragment from row-major f16 LDS tile (row stride = 128 halves) ----
// ISA 16-bit A layout: lane l -> M = l&15, h = l>>4; halves j=0..7 are K = kc*32+8h+j,
// halves j=8..15 are K = kc*32+16+8h+(j-8)  => two contiguous 16B segments.
static __device__ inline v16h load_a_frag(const _Float16* tile, int m_base, int kc, int lane) {
    int M = m_base + (lane & 15);
    int h = lane >> 4;
    const _Float16* base = tile + M * 128 + kc * 32;
    v8h lo = *(const v8h*)(base + 8 * h);
    v8h hi = *(const v8h*)(base + 16 + 8 * h);
    return combine16(lo, hi);
}

// ---- B-matrix fragment from pre-swizzled weight storage (per-lane 32B contiguous) ----
static __device__ inline v16h load_b_frag(const _Float16* frag, int nt_kc, int lane) {
    const _Float16* p = frag + ((size_t)nt_kc * 32 + lane) * 16;
    v8h lo = *(const v8h*)(p);
    v8h hi = *(const v8h*)(p + 8);
    return combine16(lo, hi);
}

// Shared K=256 GEMM: A = [X | Y] (two 16x128 f16 LDS tiles), B = fW fragments (KC=8)
static __device__ inline v8f gemm256(const _Float16* X, const _Float16* Y,
                                     const _Float16* fW, int nt, int lane) {
    v8f acc = {};
    #pragma unroll
    for (int kc = 0; kc < 8; ++kc) {
        v16h a = (kc < 4) ? load_a_frag(X, 0, kc, lane)
                          : load_a_frag(Y, 0, kc - 4, lane);
        v16h b = load_b_frag(fW, nt * 8 + kc, lane);
        acc = wmma_f16(a, b, acc);
    }
    return acc;
}

// =====================================================================
// Kernel 0: convert f32 row-major [K,N] weight to f16 B-fragment layout
// =====================================================================
__global__ void prep_weight_frag(const float* __restrict__ W, _Float16* __restrict__ dst,
                                 int K, int N) {
    int t = blockIdx.x * blockDim.x + threadIdx.x;
    int KC = K / 32;
    int total = (N / 16) * KC * 32;
    if (t >= total) return;
    int lane = t & 31;
    int kc = (t >> 5) % KC;
    int nt = t / (32 * KC);
    int n = nt * 16 + (lane & 15);
    int h = lane >> 4;
    _Float16* o = dst + (size_t)t * 16;
    #pragma unroll
    for (int j = 0; j < 16; ++j) {
        int kidx = kc * 32 + (j < 8 ? 8 * h + j : 16 + 8 * h + (j - 8));       // B-layout K index
        o[j] = (_Float16)W[(size_t)kidx * N + n];
    }
}

// =====================================================================
// Kernel 1: CT[n,:] = cur[n,:] @ W_msg_top + b_msg   (16 cells per block)
// =====================================================================
__global__ __launch_bounds__(256)
void cellterm_kernel(const float* __restrict__ cur, const _Float16* __restrict__ fWtop,
                     const float* __restrict__ b_msg, float* __restrict__ CT) {
    __shared__ __align__(16) _Float16 A16[16 * 128];
    __shared__ __align__(16) float    CTs[16 * 128];
    int t = threadIdx.x, lane = t & 31, w = t >> 5;
    int row0 = blockIdx.x * 16;
    bool full = (row0 + 16 <= N_CELLS);        // uniform per block

    if (full) {
        for (int i = t; i < 512; i += 256) {
            int r = i >> 5, c4 = (i & 31) * 4;
            v4f v = *(const v4f*)(cur + (size_t)(row0 + r) * D + c4);
            *(v4h*)&A16[r * 128 + c4] = __builtin_convertvector(v, v4h);
        }
    } else {
        for (int i = t; i < 512; i += 256) {
            int r = i >> 5, c4 = (i & 31) * 4;
            int g = row0 + r; if (g >= N_CELLS) g = N_CELLS - 1;
            v4f v = *(const v4f*)(cur + (size_t)g * D + c4);
            *(v4h*)&A16[r * 128 + c4] = __builtin_convertvector(v, v4h);
        }
    }
    __syncthreads();

    v8f acc = {};
    #pragma unroll
    for (int kc = 0; kc < 4; ++kc) {
        v16h a = load_a_frag(A16, 0, kc, lane);
        v16h b = load_b_frag(fWtop, w * 4 + kc, lane);
        acc = wmma_f16(a, b, acc);
    }
    int n = w * 16 + (lane & 15);
    float bn = b_msg[n];
    #pragma unroll
    for (int v = 0; v < 8; ++v) {
        int M = v + 8 * (lane >> 4);
        CTs[M * 128 + n] = acc[v] + bn;
    }
    __syncthreads();

    if (full) {
        for (int i = t; i < 512; i += 256) {
            int r = i >> 5, c4 = (i & 31) * 4;
            *(v4f*)(CT + (size_t)(row0 + r) * D + c4) = *(const v4f*)&CTs[r * 128 + c4];
        }
    } else {
        for (int i = t; i < 512; i += 256) {
            int r = i >> 5, c4 = (i & 31) * 4;
            if (row0 + r < N_CELLS)
                *(v4f*)(CT + (size_t)(row0 + r) * D + c4) = *(const v4f*)&CTs[r * 128 + c4];
        }
    }
}

// =====================================================================
// Kernel 2: per-cell edge processing (one workgroup per cell)
//   agg_l, agg_d = masked means of raw neighbors
//   agg_msg     = masked mean of relu(CT + nbr_k @ W_msg_bot)
// =====================================================================
__global__ __launch_bounds__(256)
void edge_kernel(const float* __restrict__ nbr, const int* __restrict__ conn,
                 const _Float16* __restrict__ fWbot, const float* __restrict__ CT,
                 float* __restrict__ aggL, float* __restrict__ aggM,
                 float* __restrict__ aggD) {
    __shared__ __align__(16) _Float16 A16[32 * 128];     // padded edge rows, f16
    __shared__ float maskL[32], maskF[32], maskD[32];
    __shared__ float cnts[3];
    __shared__ __align__(16) float partL[8][128];
    __shared__ __align__(16) float partD[8][128];
    __shared__ float ct_s[128];
    int cell = blockIdx.x;
    int t = threadIdx.x, lane = t & 31, w = t >> 5;
    const float* base = nbr + (size_t)cell * KNBR * D;

    // start the 13KB neighbor stream early (one cacheline per thread)
    if (t < 104) __builtin_prefetch((const char*)base + t * 128, 0, 3);

    if (t < 32) {
        int ctv = (t < KNBR) ? conn[(size_t)cell * KNBR + t] : -1;
        maskL[t] = (ctv == 0) ? 1.f : 0.f;
        maskF[t] = (ctv == 1) ? 1.f : 0.f;
        maskD[t] = (ctv == 2) ? 1.f : 0.f;
    }
    if (t < 128) ct_s[t] = CT[(size_t)cell * D + t];
    // zero pad rows 26..31
    for (int i = t; i < 6 * 128; i += 256) A16[26 * 128 + i] = (_Float16)0.f;
    __syncthreads();

    if (t == 0) {
        float cl = 0, cf = 0, cd = 0;
        for (int k = 0; k < KNBR; ++k) { cl += maskL[k]; cf += maskF[k]; cd += maskD[k]; }
        cnts[0] = fmaxf(cl, 1.f); cnts[1] = fmaxf(cf, 1.f); cnts[2] = fmaxf(cd, 1.f);
    }

    // vectorized neighbor streaming: lane owns 4 channels, wave w owns rows {w, w+8, w+16, w+24}
    {
        int d4 = lane * 4;
        v4f sl = {0.f, 0.f, 0.f, 0.f}, sd = {0.f, 0.f, 0.f, 0.f};
        #pragma unroll
        for (int kk = 0; kk < 4; ++kk) {
            int k = w + kk * 8;
            if (k < KNBR) {
                v4f v = *(const v4f*)(base + (size_t)k * D + d4);
                *(v4h*)&A16[k * 128 + d4] = __builtin_convertvector(v, v4h);
                sl += maskL[k] * v;
                sd += maskD[k] * v;
            }
        }
        *(v4f*)&partL[w][d4] = sl;
        *(v4f*)&partD[w][d4] = sd;
    }
    __syncthreads();

    if (t < 128) {
        float sl = 0.f, sd = 0.f;
        #pragma unroll
        for (int r = 0; r < 8; ++r) { sl += partL[r][t]; sd += partD[r][t]; }
        aggL[(size_t)cell * D + t] = sl / cnts[0];
        aggD[(size_t)cell * D + t] = sd / cnts[2];
    }

    // 32x128x128 GEMM: wave w owns N-tile w, both M-tiles
    v8f c0 = {}, c1 = {};
    #pragma unroll
    for (int kc = 0; kc < 4; ++kc) {
        v16h b  = load_b_frag(fWbot, w * 4 + kc, lane);
        v16h a0 = load_a_frag(A16, 0, kc, lane);
        v16h a1 = load_a_frag(A16, 16, kc, lane);
        c0 = wmma_f16(a0, b, c0);
        c1 = wmma_f16(a1, b, c1);
    }

    // epilogue: relu(CT + edge) * maskF, reduce over all 32 rows
    int n = w * 16 + (lane & 15);
    float ctn = ct_s[n];
    float p = 0.f;
    #pragma unroll
    for (int v = 0; v < 8; ++v) {
        int M0 = v + 8 * (lane >> 4);
        p += maskF[M0]      * fmaxf(c0[v] + ctn, 0.f);
        p += maskF[M0 + 16] * fmaxf(c1[v] + ctn, 0.f);
    }
    p += __shfl_xor(p, 16, 32);               // fold lanes l <-> l+16 (M halves)
    if ((lane >> 4) == 0)
        aggM[(size_t)cell * D + n] = p / cnts[1];
}

// =====================================================================
// Kernel 3: per-16-cell tile — experts, CNF, gating, combine
// =====================================================================
__global__ __launch_bounds__(256)
void final_kernel(const float* __restrict__ cur,
                  const float* __restrict__ aggL, const float* __restrict__ aggM,
                  const float* __restrict__ aggD,
                  const _Float16* __restrict__ fWl, const _Float16* __restrict__ fWu,
                  const _Float16* __restrict__ fWc, const _Float16* __restrict__ fWg1,
                  const float* __restrict__ bl, const float* __restrict__ bu,
                  const float* __restrict__ bc, const float* __restrict__ bg1,
                  const float* __restrict__ bg2, const float* __restrict__ Wg2,
                  float* __restrict__ out) {
    __shared__ __align__(16) _Float16 cur16[16 * 128];
    __shared__ __align__(16) _Float16 aL16[16 * 128];
    __shared__ __align__(16) _Float16 aM16[16 * 128];
    __shared__ __align__(16) _Float16 aD16[16 * 128];
    __shared__ __align__(16) _Float16 s16[16 * 128];
    __shared__ __align__(16) float sF[16 * 128];
    __shared__ __align__(16) float outF[16 * 128];
    __shared__ __align__(16) _Float16 h16[16 * 64];
    __shared__ float gates[16][3];
    int t = threadIdx.x, lane = t & 31, w = t >> 5;
    int row0 = blockIdx.x * 16;
    bool full = (row0 + 16 <= N_CELLS);        // uniform per block

    if (full) {
        for (int i = t; i < 512; i += 256) {
            int r = i >> 5, c4 = (i & 31) * 4;
            size_t idx = (size_t)(row0 + r) * D + c4;
            int o = r * 128 + c4;
            v4f cv = *(const v4f*)(cur + idx);
            v4h ch = __builtin_convertvector(cv, v4h);
            *(v4h*)&cur16[o] = ch;  *(v4h*)&s16[o] = ch;  *(v4f*)&sF[o] = cv;
            *(v4h*)&aL16[o] = __builtin_convertvector(*(const v4f*)(aggL + idx), v4h);
            *(v4h*)&aM16[o] = __builtin_convertvector(*(const v4f*)(aggM + idx), v4h);
            *(v4h*)&aD16[o] = __builtin_convertvector(*(const v4f*)(aggD + idx), v4h);
        }
    } else {
        for (int i = t; i < 512; i += 256) {
            int r = i >> 5, c4 = (i & 31) * 4;
            int g = row0 + r; if (g >= N_CELLS) g = N_CELLS - 1;
            size_t idx = (size_t)g * D + c4;
            int o = r * 128 + c4;
            v4f cv = *(const v4f*)(cur + idx);
            v4h ch = __builtin_convertvector(cv, v4h);
            *(v4h*)&cur16[o] = ch;  *(v4h*)&s16[o] = ch;  *(v4f*)&sF[o] = cv;
            *(v4h*)&aL16[o] = __builtin_convertvector(*(const v4f*)(aggL + idx), v4h);
            *(v4h*)&aM16[o] = __builtin_convertvector(*(const v4f*)(aggM + idx), v4h);
            *(v4h*)&aD16[o] = __builtin_convertvector(*(const v4f*)(aggD + idx), v4h);
        }
    }
    __syncthreads();

    int n = w * 16 + (lane & 15);

    // local expert: tanh([cur|agg_l] @ W_local + b_local)
    v8f cl = gemm256(cur16, aL16, fWl, w, lane);
    float bln = bl[n];
    float localv[8];
    #pragma unroll
    for (int v = 0; v < 8; ++v) localv[v] = tanhf(cl[v] + bln);

    // functional expert: tanh([cur|agg_msg] @ W_upd + b_upd)
    v8f cf = gemm256(cur16, aM16, fWu, w, lane);
    float bun = bu[n];
    float funcv[8];
    #pragma unroll
    for (int v = 0; v < 8; ++v) funcv[v] = tanhf(cf[v] + bun);

    // gating layer 1 (waves 0..3, N=64): h = relu(cur @ W_g1 + b_g1)
    if (w < 4) {
        v8f cg = {};
        #pragma unroll
        for (int kc = 0; kc < 4; ++kc) {
            v16h a = load_a_frag(cur16, 0, kc, lane);
            v16h b = load_b_frag(fWg1, w * 4 + kc, lane);
            cg = wmma_f16(a, b, cg);
        }
        int ng = w * 16 + (lane & 15);
        float bg = bg1[ng];
        #pragma unroll
        for (int v = 0; v < 8; ++v) {
            int M = v + 8 * (lane >> 4);
            h16[M * 64 + ng] = (_Float16)fmaxf(cg[v] + bg, 0.f);
        }
    }

    // CNF: 3 Euler steps  s += DT * tanh([s|agg_d] @ W_cnf + b_cnf)
    float bcn = bc[n];
    for (int step = 0; step < 3; ++step) {
        v8f cc = gemm256(s16, aD16, fWc, w, lane);
        __syncthreads();                       // everyone done reading s16
        #pragma unroll
        for (int v = 0; v < 8; ++v) {
            int M = v + 8 * (lane >> 4);
            float ns = sF[M * 128 + n] + DT * tanhf(cc[v] + bcn);
            sF[M * 128 + n] = ns;
            s16[M * 128 + n] = (_Float16)ns;
        }
        __syncthreads();                       // new s16 visible
    }

    // gating layer 2 + softmax (threads 0..15, one cell each)
    if (t < 16) {
        float l0 = bg2[0], l1 = bg2[1], l2 = bg2[2];
        for (int j = 0; j < HG; ++j) {
            float hv = (float)h16[t * 64 + j];
            l0 += hv * Wg2[j * 3 + 0];
            l1 += hv * Wg2[j * 3 + 1];
            l2 += hv * Wg2[j * 3 + 2];
        }
        float mx = fmaxf(l0, fmaxf(l1, l2));
        float e0 = expf(l0 - mx), e1 = expf(l1 - mx), e2 = expf(l2 - mx);
        float s = e0 + e1 + e2;
        gates[t][0] = e0 / s; gates[t][1] = e1 / s; gates[t][2] = e2 / s;
    }
    __syncthreads();

    // gated combine -> LDS stage
    #pragma unroll
    for (int v = 0; v < 8; ++v) {
        int M = v + 8 * (lane >> 4);
        outF[M * 128 + n] = gates[M][0] * localv[v]
                          + gates[M][1] * funcv[v]
                          + gates[M][2] * sF[M * 128 + n];
    }
    __syncthreads();

    // coalesced b128 writeback
    if (full) {
        for (int i = t; i < 512; i += 256) {
            int r = i >> 5, c4 = (i & 31) * 4;
            *(v4f*)(out + (size_t)(row0 + r) * D + c4) = *(const v4f*)&outF[r * 128 + c4];
        }
    } else {
        for (int i = t; i < 512; i += 256) {
            int r = i >> 5, c4 = (i & 31) * 4;
            if (row0 + r < N_CELLS)
                *(v4f*)(out + (size_t)(row0 + r) * D + c4) = *(const v4f*)&outF[r * 128 + c4];
        }
    }
}

// =====================================================================
extern "C" void kernel_launch(void* const* d_in, const int* in_sizes, int n_in,
                              void* d_out, int out_size, void* d_ws, size_t ws_size,
                              hipStream_t stream) {
    (void)in_sizes; (void)n_in; (void)out_size; (void)ws_size;
    const float* cur     = (const float*)d_in[0];
    const float* nbr     = (const float*)d_in[1];
    const int*   conn    = (const int*)d_in[2];
    const float* W_local = (const float*)d_in[3];
    const float* b_local = (const float*)d_in[4];
    const float* W_msg   = (const float*)d_in[5];
    const float* b_msg   = (const float*)d_in[6];
    const float* W_upd   = (const float*)d_in[7];
    const float* b_upd   = (const float*)d_in[8];
    const float* W_cnf   = (const float*)d_in[9];
    const float* b_cnf   = (const float*)d_in[10];
    const float* W_g1    = (const float*)d_in[11];
    const float* b_g1    = (const float*)d_in[12];
    const float* W_g2    = (const float*)d_in[13];
    const float* b_g2    = (const float*)d_in[14];
    float* out = (float*)d_out;

    char* ws = (char*)d_ws;
    size_t NB = (size_t)N_CELLS * D * sizeof(float);
    float* CT   = (float*)(ws);
    float* aggL = (float*)(ws + NB);
    float* aggM = (float*)(ws + 2 * NB);
    float* aggD = (float*)(ws + 3 * NB);
    _Float16* fWl   = (_Float16*)(ws + 4 * NB);
    _Float16* fWtop = fWl   + 256 * 128;
    _Float16* fWbot = fWtop + 128 * 128;
    _Float16* fWu   = fWbot + 128 * 128;
    _Float16* fWc   = fWu   + 256 * 128;
    _Float16* fWg1  = fWc   + 256 * 128;

    // one-time weight swizzle to f16 B-fragment layout
    prep_weight_frag<<<8, 256, 0, stream>>>(W_local,          fWl,   256, 128);
    prep_weight_frag<<<4, 256, 0, stream>>>(W_msg,            fWtop, 128, 128);
    prep_weight_frag<<<4, 256, 0, stream>>>(W_msg + 128*128,  fWbot, 128, 128);
    prep_weight_frag<<<8, 256, 0, stream>>>(W_upd,            fWu,   256, 128);
    prep_weight_frag<<<8, 256, 0, stream>>>(W_cnf,            fWc,   256, 128);
    prep_weight_frag<<<2, 256, 0, stream>>>(W_g1,             fWg1,  128,  64);

    int tiles = (N_CELLS + 15) / 16;   // 1231
    cellterm_kernel<<<tiles, 256, 0, stream>>>(cur, fWtop, b_msg, CT);
    edge_kernel<<<N_CELLS, 256, 0, stream>>>(nbr, conn, fWbot, CT, aggL, aggM, aggD);
    final_kernel<<<tiles, 256, 0, stream>>>(cur, aggL, aggM, aggD,
                                            fWl, fWu, fWc, fWg1,
                                            b_local, b_upd, b_cnf, b_g1, b_g2, W_g2,
                                            out);
}
